// CrfRnnLayer_21981642621281
// MI455X (gfx1250) — matrix operand order repaired and checked by reading, not compiled, not run on hardware
//
#include <hip/hip_runtime.h>
#include <hip/hip_bf16.h>
#include <math.h>

typedef __attribute__((ext_vector_type(16))) _Float16 v16h;
typedef __attribute__((ext_vector_type(8)))  float    v8f;

#define HH 56
#define WW 56
#define NN (HH * WW)        // 3136
#define NC 2
#define JT 196              // NN / 16 j-tiles
#define ICK 98              // NN / 32 i-chunks
#define NTILES (JT * ICK)   // 19208
#define TILE_HALFS 512      // 32 lanes * 16 halfs per WMMA operand tile
#define NUM_ITER 10

// ---------------------------------------------------------------------------
// Build both kernel matrices, pre-swizzled into the WMMA B-operand (32x16 f16)
// register layout. Tile t = jt*ICK + ic covers K[i in ic*32..+32)[j in jt*16..+16).
// Within a tile, element e = l*16 + h maps to lane l, VGPR h/2, half h%2:
//   K_b = h + 16*(l>=16), n = l&15  (lanes 0-15 hold K rows 0-15, 16-31 hold 16-31)
// ---------------------------------------------------------------------------
__global__ void build_tiles(const float* __restrict__ in,
                            _Float16* __restrict__ bsp,
                            _Float16* __restrict__ bbl) {
  const float* gray = in + NN;  // inputs[1]
  const int t  = blockIdx.x;
  const int jt = t / ICK;
  const int ic = t % ICK;
  for (int e = threadIdx.x; e < TILE_HALFS; e += blockDim.x) {
    const int l  = e >> 4;
    const int h  = e & 15;
    const int kb = h + ((l >= 16) ? 16 : 0);
    const int n  = l & 15;
    const int i  = ic * 32 + kb;
    const int j  = jt * 16 + n;
    const float dx = (float)(i % WW) - (float)(j % WW);
    const float dy = (float)(i / WW) - (float)(j / WW);
    const float d2 = dx * dx + dy * dy;
    const float dg = (gray[i] - gray[j]) * 255.0f;
    const float ksp = __expf(-d2 * (1.0f / 18.0f));
    const float kbl = __expf(-d2 * (1.0f / 51200.0f) - dg * dg * (1.0f / 6.0f));
    const size_t off = (size_t)t * TILE_HALFS + (size_t)e;
    bsp[off] = (_Float16)ksp;
    bbl[off] = (_Float16)kbl;
  }
}

// ---------------------------------------------------------------------------
// Row sums norm[i] = sum_j K(i,j) (symmetric, so == reference K @ ones).
// Values rounded through f16 to match what the WMMA actually accumulates.
// Deterministic LDS tree reduction (no float atomics).
// ---------------------------------------------------------------------------
__global__ void row_norms(const float* __restrict__ in,
                          float* __restrict__ nsp, float* __restrict__ nbl) {
  const float* gray = in + NN;
  const int i = blockIdx.x;
  const float xi = (float)(i % WW), yi = (float)(i / WW);
  const float gi = gray[i] * 255.0f;
  float ssp = 0.0f, sbl = 0.0f;
  for (int j = threadIdx.x; j < NN; j += blockDim.x) {
    const float dx = xi - (float)(j % WW);
    const float dy = yi - (float)(j / WW);
    const float d2 = dx * dx + dy * dy;
    const float dg = gi - gray[j] * 255.0f;
    ssp += (float)(_Float16)__expf(-d2 * (1.0f / 18.0f));
    sbl += (float)(_Float16)__expf(-d2 * (1.0f / 51200.0f) - dg * dg * (1.0f / 6.0f));
  }
  __shared__ float rs[256], rb[256];
  rs[threadIdx.x] = ssp;
  rb[threadIdx.x] = sbl;
  __syncthreads();
  for (int s = 128; s > 0; s >>= 1) {
    if (threadIdx.x < s) {
      rs[threadIdx.x] += rs[threadIdx.x + s];
      rb[threadIdx.x] += rb[threadIdx.x + s];
    }
    __syncthreads();
  }
  if (threadIdx.x == 0) { nsp[i] = rs[0]; nbl[i] = rb[0]; }
}

__global__ void init_unaries(const float* __restrict__ in, float* __restrict__ un) {
  const int n = blockIdx.x * blockDim.x + threadIdx.x;
  if (n < NN) {
    const float u = in[n];        // inputs[0]
    un[n]      = 1.0f - u;        // channel 0
    un[NN + n] = u;               // channel 1
  }
}

// ---------------------------------------------------------------------------
// Build the A operand (16x32 f16, rows >= NC zero) in a contiguous-k A-fragment
// layout: lane l (M = l&15), half h -> K = h + 16*(l>=16).
// iter 0 passes q through unchanged (reference semantics), else channel softmax.
// ---------------------------------------------------------------------------
__global__ void make_afrag(const float* __restrict__ q,
                           _Float16* __restrict__ afrag, const int do_softmax) {
  const int ic = blockIdx.x;
  const int e  = threadIdx.x;       // blockDim.x == 512 == TILE_HALFS
  const int l  = e >> 4;
  const int h  = e & 15;
  const int M  = l & 15;
  const int K  = h + ((l >= 16) ? 16 : 0);
  const int i  = ic * 32 + K;
  float s = 0.0f;
  if (M < NC) {
    const float q0 = q[i];
    const float q1 = q[NN + i];
    if (do_softmax) {
      const float m  = fmaxf(q0, q1);
      const float e0 = __expf(q0 - m);
      const float e1 = __expf(q1 - m);
      s = ((M == 0) ? e0 : e1) / (e0 + e1);
    } else {
      s = (M == 0) ? q0 : q1;
    }
  }
  afrag[(size_t)ic * TILE_HALFS + (size_t)e] = (_Float16)s;
}

// ---------------------------------------------------------------------------
// One mean-field iteration. 196 blocks (one per 16-column j-tile), 8 waves.
// Each wave WMMA-accumulates a strided subset of the 98 i-chunks for both
// kernel matrices. The wave index is pinned to an SGPR via readfirstlane so
// the chunk loop is scalar (s_cmp/s_cbranch, no EXEC masking): EXEC is
// provably all-1s at every WMMA as the ISA requires. The 2 live accumulator
// rows (c=0 -> VGPR0 lanes0-15, c=1 -> VGPR1 lanes0-15) are reduced across
// waves in LDS, then normalized and mixed by the three 2x2 weight matrices.
// ---------------------------------------------------------------------------
__global__ void __launch_bounds__(256)
crf_iter(const _Float16* __restrict__ bsp, const _Float16* __restrict__ bbl,
         const _Float16* __restrict__ afrag,
         const float* __restrict__ nsp, const float* __restrict__ nbl,
         const float* __restrict__ un,
         const float* __restrict__ spw, const float* __restrict__ blw,
         const float* __restrict__ cmp,
         float* __restrict__ qout) {
  const int jt   = blockIdx.x;
  const int tid  = threadIdx.x;
  const int w    = __builtin_amdgcn_readfirstlane(tid >> 5);  // wave-uniform SGPR
  const int lane = tid & 31;

  v8f csp = {};
  v8f cbl = {};
  // Running per-wave tile pointers; each tile is 32 v16h (1 KB), lane owns one.
  const v16h* Ap = (const v16h*)afrag + (size_t)w * 32 + lane;
  const v16h* Bs = (const v16h*)bsp + ((size_t)jt * ICK + w) * 32 + lane;
  const v16h* Bb = (const v16h*)bbl + ((size_t)jt * ICK + w) * 32 + lane;

  const int steps = (ICK - w + 7) >> 3;   // ceil((ICK - w) / 8), scalar
#pragma unroll 2
  for (int s = 0; s < steps; ++s) {
    const v16h a  = *Ap;
    const v16h b0 = *Bs;
    const v16h b1 = *Bb;
    csp = __builtin_amdgcn_wmma_f32_16x16x32_f16(false, a, false, b0,
                                                 (short)0, csp, false, false);
    cbl = __builtin_amdgcn_wmma_f32_16x16x32_f16(false, a, false, b1,
                                                 (short)0, cbl, false, false);
    Ap += 8 * 32;
    Bs += 8 * 32;
    Bb += 8 * 32;
  }

  __shared__ float red[8][4][16];
  if (lane < 16) {
    red[w][0][lane] = csp[0];
    red[w][1][lane] = csp[1];
    red[w][2][lane] = cbl[0];
    red[w][3][lane] = cbl[1];
  }
  __syncthreads();

  if (tid < 16) {
    const int n = tid;
    float s0 = 0.0f, s1 = 0.0f, b0 = 0.0f, b1 = 0.0f;
    for (int ww = 0; ww < 8; ++ww) {
      s0 += red[ww][0][n];
      s1 += red[ww][1][n];
      b0 += red[ww][2][n];
      b1 += red[ww][3][n];
    }
    const int j = jt * 16 + n;
    const float isp = 1.0f / nsp[j];
    const float ibl = 1.0f / nbl[j];
    const float sp0 = s0 * isp, sp1 = s1 * isp;
    const float bl0 = b0 * ibl, bl1 = b1 * ibl;
    const float m0 = spw[0] * sp0 + spw[1] * sp1 + blw[0] * bl0 + blw[1] * bl1;
    const float m1 = spw[2] * sp0 + spw[3] * sp1 + blw[2] * bl0 + blw[3] * bl1;
    const float p0 = cmp[0] * m0 + cmp[1] * m1;
    const float p1 = cmp[2] * m0 + cmp[3] * m1;
    qout[j]      = un[j]      - p0;
    qout[NN + j] = un[NN + j] - p1;
  }
}

__global__ void final_softmax(const float* __restrict__ q, float* __restrict__ out) {
  const int n = blockIdx.x * blockDim.x + threadIdx.x;
  if (n < NN) {
    out[n] = 1.0f / (1.0f + __expf(q[n] - q[NN + n]));  // softmax channel 1
  }
}

// ---------------------------------------------------------------------------
extern "C" void kernel_launch(void* const* d_in, const int* in_sizes, int n_in,
                              void* d_out, int out_size, void* d_ws, size_t ws_size,
                              hipStream_t stream) {
  (void)in_sizes; (void)n_in; (void)out_size; (void)ws_size;
  const float* inp = (const float*)d_in[0];  // [2,56,56]
  const float* spw = (const float*)d_in[1];  // [2,2]
  const float* blw = (const float*)d_in[2];  // [2,2]
  const float* cmp = (const float*)d_in[3];  // [2,2]

  char* ws = (char*)d_ws;
  size_t off = 0;
  auto alloc = [&](size_t bytes) -> void* {
    void* p = ws + off;
    off = (off + bytes + 255) & ~(size_t)255;
    return p;
  };
  _Float16* bsp   = (_Float16*)alloc((size_t)NTILES * TILE_HALFS * sizeof(_Float16));
  _Float16* bbl   = (_Float16*)alloc((size_t)NTILES * TILE_HALFS * sizeof(_Float16));
  _Float16* afrag = (_Float16*)alloc((size_t)ICK * TILE_HALFS * sizeof(_Float16));
  float* nsp = (float*)alloc(NN * sizeof(float));
  float* nbl = (float*)alloc(NN * sizeof(float));
  float* un  = (float*)alloc(2 * NN * sizeof(float));
  float* qA  = (float*)alloc(2 * NN * sizeof(float));
  float* qB  = (float*)alloc(2 * NN * sizeof(float));

  build_tiles<<<NTILES, 256, 0, stream>>>(inp, bsp, bbl);
  row_norms<<<NN, 256, 0, stream>>>(inp, nsp, nbl);
  init_unaries<<<(NN + 255) / 256, 256, 0, stream>>>(inp, un);

  const float* qin = un;
  float* bufs[2] = {qA, qB};
  for (int it = 0; it < NUM_ITER; ++it) {
    make_afrag<<<ICK, TILE_HALFS, 0, stream>>>(qin, afrag, it != 0);
    float* qo = bufs[it & 1];
    crf_iter<<<JT, 256, 0, stream>>>(bsp, bbl, afrag, nsp, nbl, un,
                                     spw, blw, cmp, qo);
    qin = qo;
  }
  final_softmax<<<(NN + 255) / 256, 256, 0, stream>>>(qin, (float*)d_out);
}
